// BitLinear_67053029425862
// MI455X (gfx1250) — compile-verified, matrix operand order checked by reading, not run on hardware
//
#include <hip/hip_runtime.h>

typedef __attribute__((ext_vector_type(8))) int v8i;

#define IN_DIM   8192
#define OUT_DIM  8192
#define BATCH    16
#define QMAX     127.0f
#define EPS_Q    1e-8f

#define GEMM_WAVES   8                       // waves per workgroup (K-split)
#define KCHUNK       (IN_DIM / GEMM_WAVES)   // 1024 K per wave

// ---------------------------------------------------------------------------
// Workspace layout (d_ws):
//   float ws[0] : scale      = 127 / max(|x|, eps)
//   float ws[1] : inv_scale  = max(|x|, eps) / 127
//   float ws[2..65] : 64 per-block |x| max partials
//   byte  256.. : q_x int8 [BATCH][IN_DIM]   (128 KB)
// ---------------------------------------------------------------------------

__global__ __launch_bounds__(256) void k_absmax_partial(
    const float* __restrict__ x, float* __restrict__ partial, int n) {
  __shared__ float sm[256];
  const int tid = threadIdx.x;
  float m = 0.0f;
  for (int i = blockIdx.x * blockDim.x + tid; i < n; i += gridDim.x * blockDim.x)
    m = fmaxf(m, fabsf(x[i]));
  sm[tid] = m;
  __syncthreads();
  for (int s = 128; s > 0; s >>= 1) {
    if (tid < s) sm[tid] = fmaxf(sm[tid], sm[tid + s]);
    __syncthreads();
  }
  if (tid == 0) partial[blockIdx.x] = sm[0];
}

__global__ __launch_bounds__(32) void k_finalize_scale(
    const float* __restrict__ partial, float* __restrict__ scales) {
  const int tid = threadIdx.x;                      // one wave32
  float m = fmaxf(partial[tid], partial[tid + 32]); // 64 partials
  #pragma unroll
  for (int off = 16; off > 0; off >>= 1)
    m = fmaxf(m, __shfl_xor(m, off, 32));
  if (tid == 0) {
    const float ma = fmaxf(m, EPS_Q);
    scales[0] = QMAX / ma;   // scale
    scales[1] = ma / QMAX;   // 1/scale
  }
}

__global__ __launch_bounds__(256) void k_quant_x(
    const float* __restrict__ x, const float* __restrict__ scales,
    char* __restrict__ qx, int n4) {
  const float scale = scales[0];
  const int i = blockIdx.x * blockDim.x + threadIdx.x;
  if (i < n4) {
    const float4 v = ((const float4*)x)[i];
    char4 q;
    q.x = (char)(int)fminf(fmaxf(rintf(v.x * scale), -QMAX), QMAX);
    q.y = (char)(int)fminf(fmaxf(rintf(v.y * scale), -QMAX), QMAX);
    q.z = (char)(int)fminf(fmaxf(rintf(v.z * scale), -QMAX), QMAX);
    q.w = (char)(int)fminf(fmaxf(rintf(v.w * scale), -QMAX), QMAX);
    ((char4*)qx)[i] = q;
  }
}

// One workgroup (8 waves) per 16-column output tile; each wave owns a
// K-chunk of 1024.  Pass 1: per-chunk |w| row sums -> LDS -> full-row alpha
// (first HBM read of the tile's 512 KB of weights).  Pass 2: each wave
// re-reads its own 64 KB chunk (L2 hit), ternarizes on the fly and
// accumulates with V_WMMA_I32_16X16X64_IU8.  Epilogue: 8-way i32 partial
// reduction through LDS; the 256 threads map 1:1 onto the 16x16 tile.
__global__ __launch_bounds__(GEMM_WAVES * 32) void k_bitlinear_gemm(
    const float* __restrict__ weight, const float* __restrict__ bias,
    const char* __restrict__ qx, const float* __restrict__ scales,
    float* __restrict__ out) {
  const int tid  = threadIdx.x;
  const int wave = tid >> 5;           // 0..7
  const int lane = tid & 31;           // wave32
  const int nloc = lane & 15;          // tile column == weight row within tile
  const int hi   = lane >> 4;          // half-wave selector
  const int n    = blockIdx.x * 16 + nloc;

  __shared__ float salpha[GEMM_WAVES][16];
  __shared__ int   sacc[GEMM_WAVES * 32 * 8];

  const float* __restrict__ wrow = weight + (size_t)n * IN_DIM;
  const int kbase = wave * KCHUNK;

  // --- pass 1: per-chunk sum of |w|; lane covers half of the chunk ---
  float s = 0.0f;
  {
    const int k0 = kbase + hi * (KCHUNK / 2);
    for (int k = 0; k < KCHUNK / 2; k += 4) {
      const float4 v = *(const float4*)(wrow + k0 + k);
      s += fabsf(v.x) + fabsf(v.y) + fabsf(v.z) + fabsf(v.w);
    }
  }
  s += __shfl_xor(s, 16, 32);          // combine chunk halves
  if (hi == 0) salpha[wave][nloc] = s;
  __syncthreads();
  float total = 0.0f;
  #pragma unroll
  for (int w = 0; w < GEMM_WAVES; ++w) total += salpha[w][nloc];
  const float alpha = fmaxf(total * (1.0f / IN_DIM), EPS_Q);
  const float h = 0.5f * alpha;        // ternary threshold

  // --- pass 2: K loop over this wave's chunk, 64 per WMMA ---
  v8i acc = {};
  const char* __restrict__ arow = qx + nloc * IN_DIM + hi * 8;  // A: M = lane%16

  for (int kk = kbase; kk < kbase + KCHUNK; kk += 64) {
    // A fragment: 8-bit 16x64 layout; dword pairs (j,j+1) are 8 contiguous bytes
    v8i a;
    #pragma unroll
    for (int j2 = 0; j2 < 4; ++j2) {
      const int2 av = *(const int2*)(arow + kk + j2 * 16);
      a[2 * j2]     = av.x;
      a[2 * j2 + 1] = av.y;
    }
    // B fragment: ternarize 4 consecutive weights per dword, pack to int8x4
    v8i b;
    #pragma unroll
    for (int j = 0; j < 8; ++j) {
      const int kb = kk + 16 * (j >> 1) + 4 * (j & 1) + hi * 8;
      const float4 wv = *(const float4*)(wrow + kb);
      const int t0 = (wv.x > h) - (wv.x < -h);
      const int t1 = (wv.y > h) - (wv.y < -h);
      const int t2 = (wv.z > h) - (wv.z < -h);
      const int t3 = (wv.w > h) - (wv.w < -h);
      b[j] = (t0 & 255) | ((t1 & 255) << 8) | ((t2 & 255) << 16) | ((t3 & 255) << 24);
    }
    // D = A(int8, signed) x B(int8, signed) + C(i32)
    acc = __builtin_amdgcn_wmma_i32_16x16x64_iu8(true, a, true, b, acc,
                                                 false, false);
  }

  // --- cross-wave i32 reduction through LDS ---
  {
    int* dst = &sacc[(wave * 32 + lane) * 8];
    ((int4*)dst)[0] = make_int4(acc[0], acc[1], acc[2], acc[3]);
    ((int4*)dst)[1] = make_int4(acc[4], acc[5], acc[6], acc[7]);
  }
  __syncthreads();

  // --- epilogue: 256 threads <-> 16x16 tile; invert D-fragment layout ---
  {
    const int m    = tid >> 4;                    // 0..15
    const int ncol = tid & 15;                    // 0..15 (== nloc of this tid)
    const int srcl = ncol + ((m >= 8) ? 16 : 0);  // lane holding (m, ncol)
    const int jj   = m & 7;                       // vgpr index within fragment
    int sum = 0;
    #pragma unroll
    for (int w = 0; w < GEMM_WAVES; ++w)
      sum += sacc[(w * 32 + srcl) * 8 + jj];
    const int   ng = blockIdx.x * 16 + ncol;
    const float sf = scales[1] * alpha;           // inv_scale * alpha_{ncol}
    out[(size_t)m * OUT_DIM + ng] = (float)sum * sf + bias[ng];
  }
}

extern "C" void kernel_launch(void* const* d_in, const int* in_sizes, int n_in,
                              void* d_out, int out_size, void* d_ws, size_t ws_size,
                              hipStream_t stream) {
  const float* x      = (const float*)d_in[0];   // [16, 8192]
  const float* weight = (const float*)d_in[1];   // [8192, 8192]
  const float* bias   = (const float*)d_in[2];   // [8192]
  float* out = (float*)d_out;                    // [16, 8192]

  float* scales  = (float*)d_ws;                 // 2 floats
  float* partial = (float*)d_ws + 2;             // 64 floats
  char*  qx      = (char*)d_ws + 256;            // 128 KB int8

  const int n = BATCH * IN_DIM;                  // 131072

  k_absmax_partial<<<64, 256, 0, stream>>>(x, partial, n);
  k_finalize_scale<<<1, 32, 0, stream>>>(partial, scales);
  k_quant_x<<<(n / 4 + 255) / 256, 256, 0, stream>>>(x, scales, qx, n / 4);
  k_bitlinear_gemm<<<OUT_DIM / 16, GEMM_WAVES * 32, 0, stream>>>(
      weight, bias, qx, scales, out);
}
